// LSTMBasedEBM_74096775791362
// MI455X (gfx1250) — matrix operand order repaired and checked by reading, not compiled, not run on hardware
//
#include <hip/hip_runtime.h>
#include <math.h>

// ---------------------------------------------------------------------------
// LSTM-based EBM for MI455X (gfx1250), compile-only target.
//   INPUT_SIZE=256, HIDDEN_SIZE=512, BATCH=32, SEQ_LEN=4096
// Strategy:
//   Kernel 1: pack [W_ih | W_hh] -> bf16 Wpack[2048][768]  (column of B contiguous)
//   Kernel 2: persistent single-workgroup scan (32 waves):
//     - TDM double-buffered prefetch of x_t (32x256 f32) into LDS
//     - per step: gates(32x2048) = [x_t | h] (32x768) @ Wpack^T via
//       v_wmma_f32_16x16x32_bf16 ; f32 accumulate ; W streamed from L2
//     - per-wave (i,f,g,o) quadruple ownership -> c in VGPRs, h -> LDS bf16
//     - readout after the scan
// ---------------------------------------------------------------------------

#define IN_SZ   256
#define HID     512
#define BATCH   32
#define SEQLEN  4096
#define KDIM    (IN_SZ + HID)      // 768
#define GDIM    (4 * HID)          // 2048

typedef __attribute__((ext_vector_type(16))) __bf16 v16bf;
typedef __attribute__((ext_vector_type(8)))  __bf16 v8bf;
typedef __attribute__((ext_vector_type(8)))  float  v8f;
typedef __attribute__((ext_vector_type(4)))  unsigned int u32x4;
typedef __attribute__((ext_vector_type(8)))  int    i32x8;
typedef __attribute__((ext_vector_type(4)))  int    i32x4;

__device__ __forceinline__ float sigmoidf_(float x) {
    return 1.0f / (1.0f + __expf(-x));
}

// ---------------------------------------------------------------------------
// Kernel 1: pack weights to bf16, K-concatenated, row-per-output-column.
// Wpack[n][k] : k<256 -> W_ih[n][k] ; k>=256 -> W_hh[n][k-256]
// ---------------------------------------------------------------------------
__global__ void pack_w_kernel(const float* __restrict__ Wih,
                              const float* __restrict__ Whh,
                              __bf16* __restrict__ Wpack) {
    const int total = GDIM * KDIM;
    for (int idx = blockIdx.x * blockDim.x + threadIdx.x; idx < total;
         idx += gridDim.x * blockDim.x) {
        const int n = idx / KDIM;
        const int k = idx - n * KDIM;
        const float v = (k < IN_SZ) ? Wih[n * IN_SZ + k]
                                    : Whh[n * HID + (k - IN_SZ)];
        Wpack[idx] = (__bf16)v;
    }
}

// ---------------------------------------------------------------------------
// TDM: DMA one timestep tile x[:, t, :] (32 rows x 256 f32, row stride T*I)
// into LDS at byte offset lds_off. 2D descriptor per CDNA5 ISA.
// ---------------------------------------------------------------------------
__device__ __forceinline__ void tdm_load_xt(const float* __restrict__ xg,
                                            int t, unsigned lds_off) {
#if __has_builtin(__builtin_amdgcn_tensor_load_to_lds)
    const unsigned long long ga =
        (unsigned long long)(const void*)(xg + (unsigned long long)t * IN_SZ);

    u32x4 g0;
    g0[0] = 1u;                                   // count=1, user descriptor
    g0[1] = lds_off;                              // lds_addr (bytes)
    g0[2] = (unsigned)(ga & 0xFFFFFFFFull);       // global_addr[31:0]
    g0[3] = (unsigned)((ga >> 32) & 0x01FFFFFFull) | (2u << 30); // addr hi | type=2

    i32x8 g1 = {};
    g1[0] = (int)(2u << 16);                      // data_size=2 (4 bytes)
    g1[1] = (int)((unsigned)IN_SZ << 16);         // tensor_dim0[15:0] @bits63:48
    g1[2] = (int)((unsigned)BATCH << 16);         // dim0 hi=0 | tensor_dim1 lo
    g1[3] = (int)((unsigned)IN_SZ << 16);         // dim1 hi=0 | tile_dim0=256
    g1[4] = (int)BATCH;                           // tile_dim1=32, tile_dim2=0
    g1[5] = (int)(SEQLEN * IN_SZ);                // tensor_dim0_stride (elems)
    g1[6] = 0;
    g1[7] = 0;

    i32x4 z4 = {};
    i32x8 z8 = {};
    __builtin_amdgcn_tensor_load_to_lds(g0, g1, z4, z4, z8, 0);
#else
    (void)xg; (void)t; (void)lds_off;
#endif
}

// ---------------------------------------------------------------------------
// Kernel 2: persistent scan. One workgroup, 1024 threads = 32 waves.
// Wave w: mtile = w>>4 (16 batch rows), jpair = (w&15)*2 -> owns h-column
// tiles jpair, jpair+1 and their four gate tiles each (i,f,g,o).
// ---------------------------------------------------------------------------
__global__ void __launch_bounds__(1024)
lstm_scan_kernel(const float* __restrict__ xg,
                 const __bf16* __restrict__ Wpack,
                 const float* __restrict__ bias,
                 const float* __restrict__ Wout,
                 const float* __restrict__ bout,
                 float* __restrict__ out) {
    __shared__ float  xb[2][BATCH * IN_SZ];   // 2 x 32KB f32 x_t staging (TDM dst)
    __shared__ __bf16 hb[BATCH * HID];        // 32KB bf16 hidden state

    const int tid   = threadIdx.x;
    const int lane  = tid & 31;
    const int wid   = tid >> 5;
    const int mtile = wid >> 4;         // 0..1  (batch rows 16*mtile..+15)
    const int jpair = (wid & 15) * 2;   // h-column tile base (0..30)
    const int lm    = lane & 15;        // M / N within tile
    const int lh    = lane >> 4;        // lane half

    // init hidden state to zero
    for (int i = tid; i < BATCH * HID; i += 1024) hb[i] = (__bf16)0.0f;

    // preload per-lane gate biases: column n = (jpair+jt+q*32)*16 + lm
    float breg[2][4];
#pragma unroll
    for (int jt = 0; jt < 2; ++jt)
#pragma unroll
        for (int q = 0; q < 4; ++q)
            breg[jt][q] = bias[(jpair + jt + q * 32) * 16 + lm];

    // persistent cell state in VGPRs (C/D layout rows)
    float cst[2][8];
#pragma unroll
    for (int jt = 0; jt < 2; ++jt)
#pragma unroll
        for (int r = 0; r < 8; ++r) cst[jt][r] = 0.0f;

    // prologue: DMA x_0 into buffer 0
    if (wid == 0) {
        tdm_load_xt(xg, 0, (unsigned)(unsigned long long)(uintptr_t)&xb[0][0]);
#if __has_builtin(__builtin_amdgcn_s_wait_tensorcnt)
        __builtin_amdgcn_s_wait_tensorcnt(0);
#endif
    }
    __syncthreads();

    // Opaque, loop-variant *offset* (not pointer!): defeats LICM on the
    // weight stream so the 192 B-tile loads stay inside the t-loop (no
    // scratch spills), while the Wpack base keeps its global-addrspace
    // provenance so loads lower to global_load_b128 (LOADcnt only, no
    // DScnt coupling like flat_load).
    unsigned long long woff = 0;

    for (int t = 0; t < SEQLEN; ++t) {
        const int buf = t & 1;

        asm volatile("" : "+s"(woff));
        const __bf16* wbase = Wpack + woff;

        // async prefetch x_{t+1} into the other buffer (overlaps WMMA chain)
        if (wid == 0 && (t + 1) < SEQLEN) {
            tdm_load_xt(xg, t + 1,
                        (unsigned)(unsigned long long)(uintptr_t)&xb[buf ^ 1][0]);
        }

        v8f acc[2][4];
#pragma unroll
        for (int jt = 0; jt < 2; ++jt)
#pragma unroll
            for (int q = 0; q < 4; ++q) acc[jt][q] = (v8f){};

        // gates += [x_t | h_{t-1}] @ Wpack^T, K = 768 in slices of 32
        for (int kb = 0; kb < KDIM; kb += 32) {
            // ---- A operand: 16x32 bf16, contiguous k order used by this code (k = kc + e):
            // lanes 0-15: M=lane, K in {kc..kc+15}, kc=0
            // lanes 16-31: same M set, kc=16
            v16bf a;
            const int kc = lh * 16;
            if (kb < IN_SZ) {
                const float* p = &xb[buf][(mtile * 16 + lm) * IN_SZ + kb];
#pragma unroll
                for (int e = 0; e < 8; ++e) {
                    a[e]     = (__bf16)p[kc + e];
                    a[8 + e] = (__bf16)p[8 + kc + e];
                }
            } else {
                const __bf16* p = &hb[(mtile * 16 + lm) * HID + (kb - IN_SZ)];
                union { v16bf v; struct { v8bf lo, hi; } s; } u;
                u.s.lo = *(const v8bf*)(p + kc);
                u.s.hi = *(const v8bf*)(p + 8 + kc);
                a = u.v;
            }

            // ---- B operands + WMMA: column N contiguous in Wpack row.
            // lanes 0-15: N=lane, K=kb..kb+15 ; lanes 16-31: N=lane-16, K=kb+16..kb+31
#pragma unroll
            for (int jt = 0; jt < 2; ++jt) {
#pragma unroll
                for (int q = 0; q < 4; ++q) {
                    const int n0 = (jpair + jt + q * 32) * 16;
                    const __bf16* wp =
                        wbase + (size_t)(n0 + lm) * KDIM + kb + lh * 16;
                    v16bf bm = *(const v16bf*)wp;
                    acc[jt][q] = __builtin_amdgcn_wmma_f32_16x16x32_bf16(
                        false, a, false, bm, (short)0, acc[jt][q], false, false);
                }
            }
        }

        // ---- nonlinearities + cell update (all wave-local)
        float hreg[2][8];
#pragma unroll
        for (int jt = 0; jt < 2; ++jt) {
#pragma unroll
            for (int r = 0; r < 8; ++r) {
                const float iv = sigmoidf_(acc[jt][0][r] + breg[jt][0]);
                const float fv = sigmoidf_(acc[jt][1][r] + breg[jt][1]);
                const float gv = tanhf(acc[jt][2][r] + breg[jt][2]);
                const float ov = sigmoidf_(acc[jt][3][r] + breg[jt][3]);
                const float cv = fv * cst[jt][r] + iv * gv;
                cst[jt][r] = cv;
                hreg[jt][r] = ov * tanhf(cv);
            }
        }

        __syncthreads();   // all reads of h_{t-1} complete

        // write h_t (C/D layout: row = mtile*16 + r + 8*lh, col = tile*16 + lm)
#pragma unroll
        for (int jt = 0; jt < 2; ++jt) {
#pragma unroll
            for (int r = 0; r < 8; ++r) {
                const int row = mtile * 16 + r + lh * 8;
                const int col = (jpair + jt) * 16 + lm;
                hb[row * HID + col] = (__bf16)hreg[jt][r];
            }
        }

        if (wid == 0 && (t + 1) < SEQLEN) {
#if __has_builtin(__builtin_amdgcn_s_wait_tensorcnt)
            __builtin_amdgcn_s_wait_tensorcnt(0);
#endif
        }
        __syncthreads();   // h_t + x_{t+1} visible to all waves
    }

    // ---- readout: energies[b] = h_final[b] . W_out + b_out
    if (tid < BATCH) {
        float s = 0.0f;
        for (int k = 0; k < HID; ++k)
            s += (float)hb[tid * HID + k] * Wout[k];
        out[tid] = s + bout[0];
    }
}

// ---------------------------------------------------------------------------
extern "C" void kernel_launch(void* const* d_in, const int* in_sizes, int n_in,
                              void* d_out, int out_size, void* d_ws, size_t ws_size,
                              hipStream_t stream) {
    (void)in_sizes; (void)n_in; (void)out_size; (void)ws_size;
    const float* x     = (const float*)d_in[0];
    const float* W_ih  = (const float*)d_in[1];
    const float* W_hh  = (const float*)d_in[2];
    const float* b     = (const float*)d_in[3];
    const float* W_out = (const float*)d_in[4];
    const float* b_out = (const float*)d_in[5];

    __bf16* Wpack = (__bf16*)d_ws;   // 2048*768*2 = 3 MB (L2-resident)

    pack_w_kernel<<<1024, 256, 0, stream>>>(W_ih, W_hh, Wpack);
    lstm_scan_kernel<<<1, 1024, 0, stream>>>(x, Wpack, b, W_out, b_out,
                                             (float*)d_out);
}
